// EnterpriseNeuralMemory_35485019800371
// MI455X (gfx1250) — matrix operand order repaired and checked by reading, not compiled, not run on hardware
//
#include <hip/hip_runtime.h>
#include <hip/hip_bf16.h>

// MI455X (gfx1250) wave32 WMMA implementation.
// Math: attn logits GEMM in bf16 WMMA (f32 acc) = 34.4 GFLOP; conv expert
// algebraically collapsed (mean-pool of a linear conv = conv of shifted chunk
// means) into a 1024x1536x512 bf16 GEMM; softmax/pool in f32.

typedef __attribute__((ext_vector_type(16))) __bf16 v16bf;
typedef __attribute__((ext_vector_type(4)))  __bf16 v4bf;
typedef __attribute__((ext_vector_type(8)))  float  v8f;

#define B_   8
#define S_   8192
#define D_   512
#define C_   64
#define NCH  128     // chunks per batch
#define HID_ 128
#define NEXP_ 3

// LDS row padding (bf16 elems / f32 elems) to avoid bank conflicts
#define XPAD 520
#define LPAD 516

// workspace byte offsets
#define WS_ROUT  0                          // 8*3 f32
#define WS_ATTNT 1024                       // 512*512 bf16   (attn_wT[n][k])
#define WS_WCAT  (WS_ATTNT + 512*512*2)     // 512*1536 bf16  (WcatT[o][h*512+i])
#define WS_FEAT  (WS_WCAT + 512*1536*2)     // 1024*1536 bf16 (mprev|avg|mnext)

// ---------------------------------------------------------------- router ----
__global__ __launch_bounds__(256)
void k_router(const float* __restrict__ x,
              const float* __restrict__ w1, const float* __restrict__ b1,
              const float* __restrict__ w2, const float* __restrict__ b2,
              float* __restrict__ rout) {
  __shared__ float xf[D_];
  __shared__ float h[HID_];
  const int b = blockIdx.x;
  const int t = threadIdx.x;
  const float* xb = x + (size_t)b * S_ * D_;
  for (int c = t; c < D_; c += 256) {
    float s = 0.f;
    for (int j = 0; j < NCH; ++j) s += xb[(size_t)(j * C_) * D_ + c];
    xf[c] = s * (1.0f / NCH);
  }
  __syncthreads();
  if (t < HID_) {
    float s = b1[t];
    for (int c = 0; c < D_; ++c) s += xf[c] * w1[c * HID_ + t];
    h[t] = fmaxf(s, 0.f);
  }
  __syncthreads();
  if (t == 0) {
    float lg[NEXP_];
    float m = -1e30f;
    for (int e = 0; e < NEXP_; ++e) {
      float s = b2[e];
      for (int j = 0; j < HID_; ++j) s += h[j] * w2[j * NEXP_ + e];
      lg[e] = s; m = fmaxf(m, s);
    }
    float den = 0.f;
    for (int e = 0; e < NEXP_; ++e) { lg[e] = __expf(lg[e] - m); den += lg[e]; }
    for (int e = 0; e < NEXP_; ++e) rout[b * NEXP_ + e] = lg[e] / den;
  }
}

// ------------------------------------------------ weight convert/transpose --
// attnT[n*512+k] = bf16(attn_w[k][n]);  wcatT[o*1536 + h*512 + i] = bf16(conv_w[o][i][h])
__global__ __launch_bounds__(256)
void k_prep(const float* __restrict__ attn_w, const float* __restrict__ conv_w,
            __bf16* __restrict__ attnT, __bf16* __restrict__ wcatT) {
  const int id = blockIdx.x * 256 + threadIdx.x;
  if (id < 512 * 512) {
    const int n = id >> 9, k = id & 511;
    attnT[id] = (__bf16)attn_w[k * 512 + n];
  } else {
    const int j = id - 512 * 512;          // < 512*1536
    const int o = j / 1536, kk = j % 1536;
    const int hh = kk >> 9, i = kk & 511;
    wcatT[j] = (__bf16)conv_w[((size_t)o * 512 + i) * 3 + hh];
  }
}

// ------------------------------------------------------------ main kernel ---
// one workgroup per (batch, chunk); 256 threads = 8 waves
__global__ __launch_bounds__(256)
void k_main(const float* __restrict__ x,
            const __bf16* __restrict__ attnT,
            const float* __restrict__ rout,
            __bf16* __restrict__ feat,
            float* __restrict__ out) {
  __shared__ __attribute__((aligned(16))) __bf16 lx[C_][XPAD];   // 66.6 KB
  __shared__ __attribute__((aligned(16))) float  llog[C_][LPAD]; // 132 KB
  __shared__ float lavg[D_];
  __shared__ float lpsum[256][4];

  const int t = threadIdx.x;
  const int bidx = blockIdx.x;          // b*128 + chunk
  const int b = bidx >> 7;
  const int chunk = bidx & 127;
  const float* xb = x + ((size_t)b * S_ + (size_t)chunk * C_) * D_;

  // ---- phase 1: chunk -> LDS bf16 + f32 partial column sums
  {
    const int col4 = t & 127;     // float4 column
    const int row0 = t >> 7;      // 0 or 1
    float4 s4 = make_float4(0.f, 0.f, 0.f, 0.f);
    #pragma unroll
    for (int i = 0; i < 32; ++i) {
      const int row = row0 + 2 * i;
      const float4 v = ((const float4*)(xb + (size_t)row * D_))[col4];
      s4.x += v.x; s4.y += v.y; s4.z += v.z; s4.w += v.w;
      v4bf pk; pk[0] = (__bf16)v.x; pk[1] = (__bf16)v.y;
               pk[2] = (__bf16)v.z; pk[3] = (__bf16)v.w;
      *(v4bf*)&lx[row][col4 * 4] = pk;
    }
    lpsum[t][0] = s4.x; lpsum[t][1] = s4.y; lpsum[t][2] = s4.z; lpsum[t][3] = s4.w;
  }
  __syncthreads();

  // ---- phase 2: reduce -> lavg (f32)
  if (t < 128) {
    #pragma unroll
    for (int q = 0; q < 4; ++q)
      lavg[t * 4 + q] = (lpsum[t][q] + lpsum[t + 128][q]) * (1.0f / C_);
  }
  __syncthreads();

  // ---- phase 3: conv features (mprev | avg | mnext) -> global bf16
  {
    __bf16* f = feat + (size_t)bidx * 1536;
    #pragma unroll
    for (int ci = 0; ci < 2; ++ci) {
      const int c = t + ci * 256;
      const float a = lavg[c];
      const float xp = (chunk > 0)   ? xb[(long)c - D_]           : 0.f;
      const float xn = (chunk < 127) ? xb[(size_t)C_ * D_ + c]    : 0.f;
      const float mprev = a + (xp - (float)lx[C_ - 1][c]) * (1.0f / C_);
      const float mnext = a + (xn - (float)lx[0][c])      * (1.0f / C_);
      f[c]        = (__bf16)mprev;
      f[512 + c]  = (__bf16)a;
      f[1024 + c] = (__bf16)mnext;
    }
  }

  // ---- phase 4: logits GEMM  [64,512] = lx(bf16) @ attn_w(bf16), f32 acc
  const int w    = t >> 5;      // wave 0..7 -> 64-col slab
  const int L    = t & 31;
  const int lrow = L & 15;
  const int lhi  = L >> 4;      // 0/1 half selector
  const int n0   = w * 64;

  v8f acc[4][4];
  #pragma unroll
  for (int r = 0; r < 4; ++r)
    #pragma unroll
    for (int c = 0; c < 4; ++c)
      acc[r][c] = (v8f){0.f, 0.f, 0.f, 0.f, 0.f, 0.f, 0.f, 0.f};

  for (int kk = 0; kk < 16; ++kk) {
    const int k0 = kk * 32;
    v16bf a[4];
    #pragma unroll
    for (int r = 0; r < 4; ++r) {
      const __bf16* p = &lx[r * 16 + lrow][k0 + lhi * 16];
      #pragma unroll
      for (int e = 0; e < 16; ++e) a[r][e] = p[e];
    }
    #pragma unroll
    for (int c = 0; c < 4; ++c) {
      const __bf16* pb = attnT + (size_t)(n0 + c * 16 + lrow) * 512 + k0 + lhi * 16;
      v16bf bf;
      #pragma unroll
      for (int e = 0; e < 16; ++e) bf[e] = pb[e];
      #pragma unroll
      for (int r = 0; r < 4; ++r)
        acc[r][c] = __builtin_amdgcn_wmma_f32_16x16x32_bf16(
            false, a[r], false, bf, (short)0, acc[r][c], false, false);
    }
  }

  // ---- phase 5: accumulators -> f32 logits in LDS
  #pragma unroll
  for (int r = 0; r < 4; ++r)
    #pragma unroll
    for (int c = 0; c < 4; ++c)
      #pragma unroll
      for (int j = 0; j < 8; ++j)
        llog[r * 16 + lhi * 8 + j][n0 + c * 16 + lrow] = acc[r][c][j];
  __syncthreads();

  // ---- phase 6: per-column softmax over 64 positions + pool + mix
  const float r0 = rout[b * 3 + 0], r1 = rout[b * 3 + 1];
  #pragma unroll
  for (int ci = 0; ci < 2; ++ci) {
    const int c = t + ci * 256;
    float m = -1e30f;
    for (int r = 0; r < C_; ++r) m = fmaxf(m, llog[r][c]);
    float s = 0.f, p = 0.f;
    for (int r = 0; r < C_; ++r) {
      const float e = __expf(llog[r][c] - m);
      s += e;
      p += e * (float)lx[r][c];
    }
    out[(size_t)bidx * D_ + c] = r0 * lavg[c] + r1 * (p / s);
  }
}

// ------------------------------------------- conv GEMM + final accumulate ---
// [1024,1536] @ [1536,512] bf16 WMMA; out += r2 * (gemm + conv_b)
__global__ __launch_bounds__(256)
void k_conv(const __bf16* __restrict__ feat,
            const __bf16* __restrict__ wcatT,
            const float* __restrict__ conv_b,
            const float* __restrict__ rout,
            float* __restrict__ out) {
  const int t = threadIdx.x;
  const int w = t >> 5, L = t & 31, lrow = L & 15, lhi = L >> 4;
  const int rg = blockIdx.x >> 2, cg = blockIdx.x & 3;
  const int R0 = rg * 64;
  const int ncol = cg * 128 + w * 16;

  v8f acc[4];
  #pragma unroll
  for (int r = 0; r < 4; ++r)
    acc[r] = (v8f){0.f, 0.f, 0.f, 0.f, 0.f, 0.f, 0.f, 0.f};

  for (int kk = 0; kk < 48; ++kk) {
    const int k0 = kk * 32;
    v16bf bf;
    const __bf16* pb = wcatT + (size_t)(ncol + lrow) * 1536 + k0 + lhi * 16;
    #pragma unroll
    for (int e = 0; e < 16; ++e) bf[e] = pb[e];
    #pragma unroll
    for (int r = 0; r < 4; ++r) {
      const __bf16* pa = feat + (size_t)(R0 + r * 16 + lrow) * 1536 + k0 + lhi * 16;
      v16bf a;
      #pragma unroll
      for (int e = 0; e < 16; ++e) a[e] = pa[e];
      acc[r] = __builtin_amdgcn_wmma_f32_16x16x32_bf16(
          false, a, false, bf, (short)0, acc[r], false, false);
    }
  }
  #pragma unroll
  for (int r = 0; r < 4; ++r) {
    const int rowbase = R0 + r * 16 + lhi * 8;
    const int col = ncol + lrow;
    #pragma unroll
    for (int j = 0; j < 8; ++j) {
      const int row = rowbase + j;                 // global chunk row
      const float r2 = rout[(row >> 7) * 3 + 2];
      const size_t idx = (size_t)row * D_ + col;
      out[idx] += r2 * (acc[r][j] + conv_b[col]);
    }
  }
}

// ------------------------------------------------------------------ launch --
extern "C" void kernel_launch(void* const* d_in, const int* in_sizes, int n_in,
                              void* d_out, int out_size, void* d_ws, size_t ws_size,
                              hipStream_t stream) {
  (void)in_sizes; (void)n_in; (void)out_size; (void)ws_size;
  const float* x       = (const float*)d_in[0];
  const float* attn_w  = (const float*)d_in[1];
  // d_in[2] = attn_b: constant along softmax axis -> cancels, unused
  const float* conv_w  = (const float*)d_in[3];
  const float* conv_b  = (const float*)d_in[4];
  const float* rw1     = (const float*)d_in[5];
  const float* rb1     = (const float*)d_in[6];
  const float* rw2     = (const float*)d_in[7];
  const float* rb2     = (const float*)d_in[8];
  float* out = (float*)d_out;

  char* ws = (char*)d_ws;
  float*  rout  = (float*)(ws + WS_ROUT);
  __bf16* attnT = (__bf16*)(ws + WS_ATTNT);
  __bf16* wcatT = (__bf16*)(ws + WS_WCAT);
  __bf16* feat  = (__bf16*)(ws + WS_FEAT);   // needs ~5.25 MB total ws

  k_router<<<B_, 256, 0, stream>>>(x, rw1, rb1, rw2, rb2, rout);
  k_prep<<<(512 * 512 + 512 * 1536) / 256, 256, 0, stream>>>(attn_w, conv_w, attnT, wcatT);
  k_main<<<B_ * NCH, 256, 0, stream>>>(x, attnT, rout, feat, out);
  k_conv<<<64, 256, 0, stream>>>(feat, wcatT, conv_b, rout, out);
}
